// Attention_90804198572343
// MI455X (gfx1250) — matrix operand order repaired and checked by reading, not compiled, not run on hardware
//
#include <hip/hip_runtime.h>
#include <math.h>

#define DEVI __device__ __forceinline__

typedef __bf16 bf16_t;
typedef __attribute__((ext_vector_type(16))) __bf16 v16bf;
typedef __attribute__((ext_vector_type(8)))  float  v8f;

constexpr int BB  = 2;
constexpr int TT  = 1024;
constexpr int DM  = 768;
constexpr int NH  = 12;
constexpr int NBR = 4;
constexpr int DH  = 64;
constexpr int HT  = 48;

union BF16Frag { v16bf v; bf16_t b[16]; };

DEVI v8f zero8() {
  v8f z;
#pragma unroll
  for (int i = 0; i < 8; ++i) z[i] = 0.f;
  return z;
}
// A-fragment (16x32 bf16): lane half selects K runs [K0+half*8,+8) and [K0+16+half*8,+8)
DEVI v16bf frag_f32_runs(const float* p0, const float* p1) {
  BF16Frag f;
#pragma unroll
  for (int i = 0; i < 8; ++i) { f.b[i] = (bf16_t)p0[i]; f.b[8+i] = (bf16_t)p1[i]; }
  return f.v;
}
DEVI v16bf frag_bf_runs(const bf16_t* p0, const bf16_t* p1) {
  BF16Frag f;
#pragma unroll
  for (int i = 0; i < 8; ++i) { f.b[i] = p0[i]; f.b[8+i] = p1[i]; }
  return f.v;
}
// B-fragment (32x16 bf16): lane half selects contiguous K run [K0+half*16,+16)
DEVI v16bf frag_bf_16(const bf16_t* p) {
  BF16Frag f;
#pragma unroll
  for (int i = 0; i < 16; ++i) f.b[i] = p[i];
  return f.v;
}
DEVI v8f wmma_bf16(v16bf a, v16bf b, v8f c) {
  return __builtin_amdgcn_wmma_f32_16x16x32_bf16(false, a, false, b, (short)0, c,
                                                 false, false);
}

// ---------------------------------------------------------------------------
// K0a: elementwise f32 -> bf16 (weight pre-pass; n % 4 == 0)
// ---------------------------------------------------------------------------
__global__ __launch_bounds__(256)
void cvt_bf16_kernel(const float* __restrict__ src, bf16_t* __restrict__ dst,
                     int n) {
  int i = (blockIdx.x * 256 + threadIdx.x) * 4;
  if (i < n) {
    float4 v = *(const float4*)(src + i);
    dst[i + 0] = (bf16_t)v.x;
    dst[i + 1] = (bf16_t)v.y;
    dst[i + 2] = (bf16_t)v.z;
    dst[i + 3] = (bf16_t)v.w;
  }
}

// ---------------------------------------------------------------------------
// K0b: WO_w[r][c][d] f32 -> WOT[r][d][c] bf16 (tiled LDS transpose)
// grid (DM/32, DM/32, NBR), block (32,8)
// ---------------------------------------------------------------------------
__global__ __launch_bounds__(256)
void transpose_wo_kernel(const float* __restrict__ WOw,
                         bf16_t* __restrict__ WOT) {
  __shared__ float tile[32][33];
  const int r  = blockIdx.z;
  const int tx = threadIdx.x, ty = threadIdx.y;
  const float* src = WOw + (size_t)r * DM * DM;
  bf16_t*      dst = WOT + (size_t)r * DM * DM;
  const int c0 = blockIdx.y * 32, d0 = blockIdx.x * 32;
#pragma unroll
  for (int i = 0; i < 4; ++i)
    tile[ty + i * 8][tx] = src[(size_t)(c0 + ty + i * 8) * DM + (d0 + tx)];
  __syncthreads();
#pragma unroll
  for (int i = 0; i < 4; ++i)
    dst[(size_t)(d0 + ty + i * 8) * DM + (c0 + tx)] =
        (bf16_t)tile[tx][ty + i * 8];
}

// ---------------------------------------------------------------------------
// K1: projection GEMM (A@W^T [+bias]) + wedge + RoPE, output bf16 [B,HT,T,DH]
// grid (T/16, HT, B), block 128 (4 waves); wave w owns output cols [16w,16w+16)
// ---------------------------------------------------------------------------
__global__ __launch_bounds__(128)
void proj_wedge_rope_kernel(const float* __restrict__ src,
                            const bf16_t* __restrict__ W,     // bf16 weights
                            const float* __restrict__ bias,   // nullable
                            const float* __restrict__ wedgeA,
                            const float* __restrict__ wedgeBias,
                            bf16_t* __restrict__ dst,
                            int qmode) {
  __shared__ float skew[DH][DH];   // 16 KB: wedge_A - wedge_A^T
  __shared__ float xs[16][DH];     // pre-wedge tile
  __shared__ float wd[16][DH];     // post-wedge tile
  const int tid  = threadIdx.x;
  const int lane = tid & 31, half = (tid >> 4) & 1, wv = tid >> 5;
  const int t0 = blockIdx.x * 16;
  const int h  = blockIdx.y;
  const int b  = blockIdx.z;

  for (int i = tid; i < DH * DH; i += 128) {
    int d = i >> 6, e = i & 63;
    skew[d][e] = wedgeA[d * DH + e] - wedgeA[e * DH + d];
  }

  const int m    = lane & 15;
  const int tok  = t0 + m;
  const float* arow = src + (size_t)(b * TT + tok) * DM;
  const int nloc  = wv * 16 + m;                      // 0..63 within head
  const int wrow0 = (qmode ? h : (h % NH)) * DH;
  const bf16_t* brow = W + (size_t)(wrow0 + nloc) * DM;

  v8f acc = zero8();
#pragma unroll 4
  for (int kc = 0; kc < DM / 32; ++kc) {
    const int K0 = kc * 32;
    if (kc + 1 < DM / 32) {
      __builtin_prefetch(arow + K0 + 32, 0, 1);
      __builtin_prefetch(brow + K0 + 32, 0, 1);
    }
    v16bf af = frag_f32_runs(arow + K0 + half * 8, arow + K0 + 16 + half * 8);
    v16bf bf = frag_bf_runs(brow + K0 + half * 8, brow + K0 + 16 + half * 8);
    acc = wmma_bf16(af, bf, acc);
  }
  const float bv = bias ? bias[wrow0 + nloc] : 0.f;
#pragma unroll
  for (int v = 0; v < 8; ++v) xs[v + 8 * half][nloc] = acc[v] + bv;
  __syncthreads();

  // wedge: out[e] = x[e]*(1 + bias_h[e]) + sum_d x[d]*skew[d][e]
  const float* wbh = wedgeBias + h * DH;
  for (int i = tid; i < 16 * DH; i += 128) {
    int row = i >> 6, e = i & 63;
    float s = xs[row][e] * (1.0f + wbh[e]);
#pragma unroll 8
    for (int d = 0; d < DH; ++d) s += xs[row][d] * skew[d][e];
    wd[row][e] = s;
  }
  __syncthreads();

  // RoPE + bf16 store
  bf16_t* drow0 = dst + (size_t)((b * HT + h) * TT + t0) * DH;
  for (int i = tid; i < 16 * DH; i += 128) {
    int row = i >> 6, c = i & 63;
    int j = c & 31;
    float x1 = wd[row][2 * j], x2 = wd[row][2 * j + 1];
    // inv_freq[j] = 10000^(-2j/64) = exp(-j * ln(1e4)/32)
    float fr = (float)(t0 + row) * __expf(-0.28782313662425572f * (float)j);
    float cs = __cosf(fr), sn = __sinf(fr);
    float y = (c < 32) ? (x1 * cs - x2 * sn) : (x1 * sn + x2 * cs);
    drow0[(size_t)row * DH + c] = (bf16_t)y;
  }
}

// ---------------------------------------------------------------------------
// K2: flash-style causal attention stats + streaming top-4 -> marker, p_sink
// grid (T/16, HT, B), block 32 (one wave owns a 16-query tile)
// ---------------------------------------------------------------------------
__global__ __launch_bounds__(32)
void attn_topk_kernel(const bf16_t* __restrict__ qb,
                      const bf16_t* __restrict__ kb,
                      const float* __restrict__ sink,
                      bf16_t* __restrict__ marker,
                      float* __restrict__ psink) {
  __shared__ float sS[16][17];
  const int lane = threadIdx.x & 31, half = (threadIdx.x >> 4) & 1;
  const int t0 = blockIdx.x * 16;
  const int h  = blockIdx.y;
  const int b  = blockIdx.z;
  const size_t bh = (size_t)(b * HT + h) * TT;
  const int m = lane & 15;

  const bf16_t* qrow = qb + (bh + (t0 + m)) * DH;
  v16bf qf0 = frag_bf_runs(qrow + half * 8,      qrow + 16 + half * 8);
  v16bf qf1 = frag_bf_runs(qrow + 32 + half * 8, qrow + 48 + half * 8);

  float mrun = -3.0e38f, Z = 0.f;
  float ts0 = -3.0e38f, ts1 = -3.0e38f, ts2 = -3.0e38f, ts3 = -3.0e38f;
  int   ti0 = 0, ti1 = 0, ti2 = 0, ti3 = 0;
  const float scale = 0.125f;   // DH^-0.5

  for (int s0 = 0; s0 <= t0; s0 += 16) {
    const bf16_t* krow = kb + (bh + (s0 + m)) * DH;
    if (s0 + 16 <= t0) __builtin_prefetch(krow + 16 * DH, 0, 1);
    v16bf kf0 = frag_bf_runs(krow + half * 8,      krow + 16 + half * 8);
    v16bf kf1 = frag_bf_runs(krow + 32 + half * 8, krow + 48 + half * 8);
    v8f acc = zero8();
    acc = wmma_bf16(qf0, kf0, acc);
    acc = wmma_bf16(qf1, kf1, acc);
#pragma unroll
    for (int v = 0; v < 8; ++v) {
      int row = v + 8 * half;              // query row within tile
      int ks  = s0 + m;                    // key index (this lane's column)
      float sc = acc[v] * scale;
      if (ks > t0 + row) sc = -3.0e38f;    // causal mask
      sS[row][m] = sc;
    }
    if (lane < 16) {                       // lane owns query row `lane`
#pragma unroll 4
      for (int n = 0; n < 16; ++n) {
        float s = sS[lane][n];
        int ks = s0 + n;
        if (s > mrun) { Z = Z * __expf(mrun - s) + 1.0f; mrun = s; }
        else          { Z += __expf(s - mrun); }
        if (s > ts0)      { ts3=ts2; ti3=ti2; ts2=ts1; ti2=ti1; ts1=ts0; ti1=ti0; ts0=s; ti0=ks; }
        else if (s > ts1) { ts3=ts2; ti3=ti2; ts2=ts1; ti2=ti1; ts1=s; ti1=ks; }
        else if (s > ts2) { ts3=ts2; ti3=ti2; ts2=s; ti2=ks; }
        else if (s > ts3) { ts3=s; ti3=ks; }
      }
    }
  }

  if (lane < 16) {
    const int qt = t0 + lane;
    float sk = sink[h];
    float M  = fmaxf(mrun, sk);
    float Ztot = Z * __expf(mrun - M) + __expf(sk - M);
    float inv  = 1.0f / Ztot;
    float pS   = __expf(sk - M) * inv;
    float v0 = __expf(ts0 - M) * inv, v1 = __expf(ts1 - M) * inv;
    float v2 = __expf(ts2 - M) * inv, v3 = __expf(ts3 - M) * inv;
    float wsum = v0 + v1 + v2 + v3 + 1e-9f;
    float w0 = v0 / wsum, w1 = v1 / wsum, w2 = v2 / wsum, w3 = v3 / wsum;
    const bf16_t* k0 = kb + (bh + ti0) * DH;
    const bf16_t* k1 = kb + (bh + ti1) * DH;
    const bf16_t* k2 = kb + (bh + ti2) * DH;
    const bf16_t* k3 = kb + (bh + ti3) * DH;
    bf16_t* mk = marker + (bh + qt) * DH;
#pragma unroll 8
    for (int d = 0; d < DH; ++d) {
      float a = w0 * (float)k0[d] + w1 * (float)k1[d] +
                w2 * (float)k2[d] + w3 * (float)k3[d];
      mk[d] = (bf16_t)a;
    }
    psink[bh + qt] = pS;
  }
}

// ---------------------------------------------------------------------------
// K3: out = gelu(marker@V1^T+b1)@V2^T + b2 + p_sink*v_null -> ctx [B,4,T,768]
// grid (T/16, HT, B), block 128 (4 waves)
// ---------------------------------------------------------------------------
__global__ __launch_bounds__(128)
void mlp_sink_kernel(const bf16_t* __restrict__ marker,
                     const float* __restrict__ psink,
                     const bf16_t* __restrict__ V1w, const float* __restrict__ V1b,
                     const bf16_t* __restrict__ V2w, const float* __restrict__ V2b,
                     const float* __restrict__ vnulls,
                     bf16_t* __restrict__ ctx) {
  __shared__ bf16_t hS[16][256];     // gelu hidden, bf16, 8 KB
  const int tid  = threadIdx.x;
  const int lane = tid & 31, half = (tid >> 4) & 1, wv = tid >> 5;
  const int t0 = blockIdx.x * 16;
  const int h  = blockIdx.y;
  const int b  = blockIdx.z;
  const size_t bh = (size_t)(b * HT + h) * TT;
  const int m = lane & 15;

  const bf16_t* arow = marker + (bh + (t0 + m)) * DH;
  v16bf af0 = frag_bf_runs(arow + half * 8,      arow + 16 + half * 8);
  v16bf af1 = frag_bf_runs(arow + 32 + half * 8, arow + 48 + half * 8);

  // phase 1: H[16x256] = gelu(marker @ V1^T + b1); wave wv -> cols [64w,64w+64)
#pragma unroll
  for (int nt = 0; nt < 4; ++nt) {
    const int j = wv * 64 + nt * 16 + m;
    const bf16_t* brow = V1w + (size_t)j * DH;
    v8f acc = zero8();
    acc = wmma_bf16(af0, frag_bf_runs(brow + half * 8,      brow + 16 + half * 8), acc);
    acc = wmma_bf16(af1, frag_bf_runs(brow + 32 + half * 8, brow + 48 + half * 8), acc);
    const float bj = V1b[j];
#pragma unroll
    for (int v = 0; v < 8; ++v) {
      float x = acc[v] + bj;
      float g = 0.5f * x * (1.0f + erff(x * 0.70710678118654752f));
      hS[v + 8 * half][j] = (bf16_t)g;
    }
  }
  __syncthreads();

  // phase 2: out[16x64] = H @ V2^T + b2; wave wv -> cols [16w,16w+16)
  const int d = wv * 16 + m;
  const bf16_t* b2row = V2w + (size_t)d * 256;
  v8f acc = zero8();
#pragma unroll
  for (int kc = 0; kc < 8; ++kc) {
    const int K0 = kc * 32;
    v16bf afr = frag_bf_runs(&hS[m][K0 + half * 8], &hS[m][K0 + 16 + half * 8]);
    v16bf bfr = frag_bf_runs(b2row + K0 + half * 8, b2row + K0 + 16 + half * 8);
    acc = wmma_bf16(afr, bfr, acc);
  }
  const int br = h / NH, nh = h % NH;
  const float vb2 = V2b[d];
  const float vnl = vnulls[br * DM + nh * DH + d];
  bf16_t* crow =
      ctx + ((size_t)(b * NBR + br) * TT + t0) * DM + nh * DH + d;
#pragma unroll
  for (int v = 0; v < 8; ++v) {
    int row = v + 8 * half;
    float pS = psink[bh + t0 + row];
    crow[(size_t)row * DM] = (bf16_t)(acc[v] + vb2 + pS * vnl);
  }
}

// ---------------------------------------------------------------------------
// K4: y = 0.25 * sum_r ctx[b,r] @ WO_w[r] + mean_r WO_b[r]
// B operand from pre-transposed bf16 WOT[r][d][c] -> contiguous fragments.
// grid (B*T/16, DM/64), block 128 (4 waves)
// ---------------------------------------------------------------------------
__global__ __launch_bounds__(128)
void out_proj_kernel(const bf16_t* __restrict__ ctx,
                     const bf16_t* __restrict__ WOT,
                     const float* __restrict__ WOb,
                     float* __restrict__ out) {
  const int tid  = threadIdx.x;
  const int lane = tid & 31, half = (tid >> 4) & 1, wv = tid >> 5;
  const int g  = blockIdx.x;
  const int b  = g >> 6;                 // T/16 == 64 tiles per batch
  const int t0 = (g & 63) * 16;
  const int m  = lane & 15;
  const int d  = blockIdx.y * 64 + wv * 16 + m;   // output channel

  v8f acc = zero8();
  for (int r = 0; r < NBR; ++r) {
    const bf16_t* arow =
        ctx + (size_t)((b * NBR + r) * TT + (t0 + m)) * DM;
    const bf16_t* brow = WOT + ((size_t)r * DM + d) * DM;   // contiguous in c
    for (int kc = 0; kc < DM / 32; ++kc) {
      const int K0 = kc * 32;
      if (kc + 1 < DM / 32) {
        __builtin_prefetch(arow + K0 + 32, 0, 1);
        __builtin_prefetch(brow + K0 + 32, 0, 1);
      }
      v16bf af = frag_bf_runs(arow + K0 + half * 8, arow + K0 + 16 + half * 8);
      v16bf bf = frag_bf_runs(brow + K0 + half * 8, brow + K0 + 16 + half * 8);
      acc = wmma_bf16(af, bf, acc);
    }
  }
  const float mb = 0.25f * (WOb[d] + WOb[DM + d] + WOb[2 * DM + d] + WOb[3 * DM + d]);
#pragma unroll
  for (int v = 0; v < 8; ++v) {
    int qt = t0 + v + 8 * half;
    out[(size_t)(b * TT + qt) * DM + d] = 0.25f * acc[v] + mb;
  }
}

// ---------------------------------------------------------------------------
extern "C" void kernel_launch(void* const* d_in, const int* in_sizes, int n_in,
                              void* d_out, int out_size, void* d_ws, size_t ws_size,
                              hipStream_t stream) {
  (void)in_sizes; (void)n_in; (void)out_size; (void)ws_size;
  const float* A    = (const float*)d_in[0];
  const float* X    = (const float*)d_in[1];
  const float* WKw  = (const float*)d_in[2];
  const float* WKb  = (const float*)d_in[3];
  const float* WQw  = (const float*)d_in[4];
  const float* wA   = (const float*)d_in[5];
  const float* wB   = (const float*)d_in[6];
  const float* sink = (const float*)d_in[7];
  const float* vnl  = (const float*)d_in[8];
  const float* V1w  = (const float*)d_in[9];
  const float* V1b  = (const float*)d_in[10];
  const float* V2w  = (const float*)d_in[11];
  const float* V2b  = (const float*)d_in[12];
  const float* WOw  = (const float*)d_in[13];
  const float* WOb  = (const float*)d_in[14];
  float* out = (float*)d_out;

  const size_t NE   = (size_t)BB * HT * TT * DH;   // 6,291,456
  const size_t NWQ  = (size_t)NBR * DM * DM;       // 2,359,296
  const size_t NWK  = (size_t)DM * DM;             //   589,824
  const size_t NV   = (size_t)4 * DH * DH;         //    16,384
  const size_t NPS  = (size_t)BB * HT * TT;        //    98,304

  bf16_t* qb  = (bf16_t*)d_ws;           // Q      (12.6 MB)
  bf16_t* kb  = qb  + NE;                // K      (12.6 MB)
  bf16_t* mk  = kb  + NE;                // marker (12.6 MB)
  float*  ps  = (float*)(mk + NE);       // p_sink ( 0.4 MB)
  bf16_t* wqb = (bf16_t*)(ps + NPS);     // WQ bf16 (4.7 MB)
  bf16_t* wkb = wqb + NWQ;               // WK bf16 (1.2 MB)
  bf16_t* v1b = wkb + NWK;               // V1 bf16
  bf16_t* v2b = v1b + NV;                // V2 bf16
  bf16_t* wot = v2b + NV;                // WO^T bf16 (4.7 MB)
  bf16_t* ctx = qb;                      // alias: Q dead after attn

  // weight pre-pass (bandwidth-trivial; runs inside the graph each call)
  cvt_bf16_kernel<<<(int)((NWQ / 4 + 255) / 256), 256, 0, stream>>>(WQw, wqb, (int)NWQ);
  cvt_bf16_kernel<<<(int)((NWK / 4 + 255) / 256), 256, 0, stream>>>(WKw, wkb, (int)NWK);
  cvt_bf16_kernel<<<(int)((NV  / 4 + 255) / 256), 256, 0, stream>>>(V1w, v1b, (int)NV);
  cvt_bf16_kernel<<<(int)((NV  / 4 + 255) / 256), 256, 0, stream>>>(V2w, v2b, (int)NV);
  transpose_wo_kernel<<<dim3(DM / 32, DM / 32, NBR), dim3(32, 8), 0, stream>>>(WOw, wot);

  dim3 g1(TT / 16, HT, BB);
  proj_wedge_rope_kernel<<<g1, 128, 0, stream>>>(A, wqb, nullptr, wA, wB, qb, 1);
  proj_wedge_rope_kernel<<<g1, 128, 0, stream>>>(X, wkb, WKb,     wA, wB, kb, 0);
  attn_topk_kernel<<<g1, 32, 0, stream>>>(qb, kb, sink, mk, ps);
  mlp_sink_kernel<<<g1, 128, 0, stream>>>(mk, ps, v1b, V1b, v2b, V2b, vnl, ctx);
  dim3 g4(BB * TT / 16, DM / 64);
  out_proj_kernel<<<g4, 128, 0, stream>>>(ctx, wot, WOb, out);
}
